// MPNN_223338300119
// MI455X (gfx1250) — compile-verified, matrix operand order checked by reading, not run on hardware
//
#include <hip/hip_runtime.h>
#include <hip/hip_bf16.h>

// ---------------- problem constants ----------------
#define NA   8        // num agents
#define NB   16384    // envs (BSZ)
#define NIN  16       // input size
#define NH_  128      // hidden dim (== embed dim)
#define POSI 2        // position column index
#define ENVB 16       // envs per block -> 128 rows per block
#define ROWS (ENVB * NA)          // 128
#define THREADS 256               // 8 waves (wave32)

typedef __attribute__((ext_vector_type(16))) _Float16 v16h;
typedef __attribute__((ext_vector_type(8)))  _Float16 v8h;
typedef __attribute__((ext_vector_type(8)))  float    v8f;
typedef __attribute__((ext_vector_type(4)))  float    v4f;

union V16U { v16h v; v8h h[2]; };

// ---- A fragment: 16x32 f16 tile from an LDS (row-major) buffer ----
// ISA 7.12.2: lanes 0-15 hold K=[0..7]&[16..23], lanes 16-31 hold K=[8..15]&[24..31]
static __device__ __forceinline__ v16h load_a_frag(const _Float16* base, int stride,
                                                   int m0, int k0, int lane) {
  int r  = m0 + (lane & 15);
  int kk = k0 + ((lane >> 4) << 3);
  V16U u;
  u.h[0] = *(const v8h*)(base + r * stride + kk);
  u.h[1] = *(const v8h*)(base + r * stride + kk + 16);
  return u.v;
}

// ---- B fragment: 32x16 tile; WT is the transposed weight (N-major, stride=K) ----
static __device__ __forceinline__ v16h load_b_frag(const _Float16* WT, int stride,
                                                   int n0, int k0, int lane) {
  int c  = n0 + (lane & 15);
  int kk = k0 + ((lane >> 4) << 3);
  V16U u;
  u.h[0] = *(const v8h*)(WT + c * stride + kk);
  u.h[1] = *(const v8h*)(WT + c * stride + kk + 16);
  return u.v;
}

// One wave computes a 16-row x 128-col GEMM tile; all shapes compile-time.
// A comes from A1 for k < SPLITK, from A2 for k >= SPLITK (concat GEMM).
// A fragments are hoisted (loaded once, reused across all 8 N-tiles).
// B fragments are loaded in clause-friendly groups of 4 before 4 back-to-back WMMAs.
template<int KDIM, int SPLITK, int STRIDEA, int STRIDEB, int STRIDED, bool BIAS, bool RELU>
static __device__ __forceinline__ void gemm_tile(
    const _Float16* __restrict__ A1, const _Float16* __restrict__ A2,
    const _Float16* __restrict__ BT, _Float16* __restrict__ D,
    const float* __restrict__ bias, int m0, int lane)
{
  constexpr int NK = KDIM / 32;
  v16h afrag[NK];
#pragma unroll
  for (int ks = 0; ks < NK; ++ks) {
    const int k0 = ks * 32;
    const _Float16* Asrc = (k0 < SPLITK) ? A1 : A2;
    const int ka = (k0 < SPLITK) ? k0 : k0 - SPLITK;
    afrag[ks] = load_a_frag(Asrc, STRIDEA, m0, ka, lane);
  }
  constexpr int CH = (NK > 4) ? 4 : NK;
#pragma unroll
  for (int nt = 0; nt < 8; ++nt) {
    const int n0 = nt * 16;
    v8f acc = {};
#pragma unroll
    for (int kb = 0; kb < NK; kb += CH) {
      v16h bfr[CH];
#pragma unroll
      for (int t = 0; t < CH; ++t)
        bfr[t] = load_b_frag(BT, STRIDEB, n0, (kb + t) * 32, lane);
#pragma unroll
      for (int t = 0; t < CH; ++t)
        acc = __builtin_amdgcn_wmma_f32_16x16x32_f16(
            /*neg_a=*/false, afrag[kb + t], /*neg_b=*/false, bfr[t],
            /*c_mod=*/(short)0, acc, /*reuse_a=*/false, /*reuse_b=*/false);
    }
    // C/D layout: lanes 0-15 -> M base 0, lanes 16-31 -> M base 8; VGPR j -> M+j; N = lane&15
    const int col   = n0 + (lane & 15);
    const int rbase = m0 + ((lane >> 4) << 3);
    const float bv  = BIAS ? bias[col] : 0.0f;
#pragma unroll
    for (int j = 0; j < 8; ++j) {
      float x = acc[j] + bv;
      if (RELU) x = fmaxf(x, 0.0f);
      D[(rbase + j) * STRIDED + col] = (_Float16)x;
    }
  }
}

// ---------------- weight prep: f32 -> f16, transposed to N-major ----------------
// ws layout (f16 elems): encT[128*32] | WqT[128*128] | WkT | WvT | WoT | updT[128*256]
__global__ void prep_weights(const float* __restrict__ encW,
                             const float* __restrict__ Wq, const float* __restrict__ Wk,
                             const float* __restrict__ Wv, const float* __restrict__ Wo,
                             const float* __restrict__ upd, _Float16* __restrict__ ws)
{
  int i = blockIdx.x * blockDim.x + threadIdx.x;
  const int ENC = 128 * 32;
  const int M2  = 128 * 128;
  if (i < ENC) {                       // encoder weight, K padded 16 -> 32
    int n = i >> 5, k = i & 31;
    ws[i] = (_Float16)((k < NIN) ? encW[k * NH_ + n] : 0.0f);
    return;
  }
  i -= ENC;
  if (i < 4 * M2) {                    // Wq/Wk/Wv/W_out, 128x128 each
    int m = i / M2, j = i % M2;
    int n = j >> 7, k = j & 127;
    const float* W = (m == 0) ? Wq : (m == 1) ? Wk : (m == 2) ? Wv : Wo;
    ws[ENC + i] = (_Float16)W[k * NH_ + n];
    return;
  }
  i -= 4 * M2;
  if (i < 128 * 256) {                 // upd_W: (256,128) -> updT (128,256)
    int n = i >> 8, k = i & 255;
    ws[ENC + 4 * M2 + i] = (_Float16)upd[k * NH_ + n];
  }
}

// ---------------- fused MPNN kernel ----------------
__global__ void __launch_bounds__(THREADS)
mpnn_fused(const float* __restrict__ inp,
           const float* __restrict__ enc_b, const float* __restrict__ upd_b,
           const _Float16* __restrict__ encT, const _Float16* __restrict__ WqT,
           const _Float16* __restrict__ WkT,  const _Float16* __restrict__ WvT,
           const _Float16* __restrict__ WoT,  const _Float16* __restrict__ updT,
           float* __restrict__ out)
{
  extern __shared__ char smem[];
  _Float16* sH0 = (_Float16*)smem;           // 128x128 f16 (h ping)
  _Float16* sH1 = sH0 + ROWS * NH_;          // 128x128 f16 (h pong)
  _Float16* sQ  = sH1 + ROWS * NH_;          // 128x128 f16 (Q; also enc staging; also m)
  _Float16* sK  = sQ  + ROWS * NH_;          // 128x128 f16 (K; also heads)
  _Float16* sV  = sK  + ROWS * NH_;          // 128x128 f16 (V)
  float* sAttn  = (float*)(sV + ROWS * NH_); // 16*8*8 scores/attn
  float* sBias  = sAttn + ENVB * 64;         // 16*8*8 mask bias (iteration-invariant)
  float* sEncB  = sBias + ENVB * 64;         // 128
  float* sUpdB  = sEncB + 128;               // 128
  float* sPos   = sUpdB + 128;               // 16 envs * 8 agents * 2

  const int tid  = threadIdx.x;
  const int lane = tid & 31;
  const int wave = tid >> 5;
  const int m0   = wave * 16;
  const int env0 = blockIdx.x * ENVB;

  // prefetch the L2-resident transposed weight panel (200 KB)
  for (int off = tid * 128; off < 204800; off += THREADS * 128)
    __builtin_prefetch(((const char*)encT) + off, 0, 1);

  if (tid < NH_) { sEncB[tid] = enc_b[tid]; sUpdB[tid] = upd_b[tid]; }

  // -------- stage positions: pos[e][a][c] --------
  {
    int e = tid >> 4, a = (tid >> 1) & 7, c = tid & 1;
    sPos[tid] = inp[((size_t)a * NB + env0 + e) * NIN + POSI + c];
  }
  __syncthreads();

  // -------- mask bias (distance > 1, diag never masked) --------
  for (int i = tid; i < ENVB * 64; i += THREADS) {
    int e = i >> 6, q = (i >> 3) & 7, g = i & 7;
    float dx = sPos[(e * 8 + q) * 2]     - sPos[(e * 8 + g) * 2];
    float dy = sPos[(e * 8 + q) * 2 + 1] - sPos[(e * 8 + g) * 2 + 1];
    sBias[i] = ((q != g) && (dx * dx + dy * dy > 1.0f)) ? -1e30f : 0.0f;
  }

  // -------- encoder staging: input (128 x 32 f16, K zero-padded 16->32) into sQ ----
  for (int i = tid; i < ROWS * 4; i += THREADS) {
    int r = i >> 2, c = (i & 3) * 4;
    int e = r >> 3, a = r & 7;
    v4f v = *(const v4f*)(inp + ((size_t)a * NB + env0 + e) * NIN + c);
    _Float16* dst = sQ + r * 32 + c;
    dst[0] = (_Float16)v.x; dst[1] = (_Float16)v.y;
    dst[2] = (_Float16)v.z; dst[3] = (_Float16)v.w;
  }
  for (int i = tid; i < ROWS * 2; i += THREADS) {
    int r = i >> 1, c = 16 + (i & 1) * 8;
    v8h z = {};
    *(v8h*)(sQ + r * 32 + c) = z;
  }
  __syncthreads();

  gemm_tile<32, 32, 32, 32, NH_, true, true>(sQ, sQ, encT, sH0, sEncB, m0, lane);
  __syncthreads();

  _Float16* hcur = sH0;
  _Float16* hnxt = sH1;

  for (int it = 0; it < 3; ++it) {
    // -------- Q, K, V projections (WMMA) --------
    gemm_tile<NH_, NH_, NH_, NH_, NH_, false, false>(hcur, hcur, WqT, sQ, nullptr, m0, lane);
    gemm_tile<NH_, NH_, NH_, NH_, NH_, false, false>(hcur, hcur, WkT, sK, nullptr, m0, lane);
    gemm_tile<NH_, NH_, NH_, NH_, NH_, false, false>(hcur, hcur, WvT, sV, nullptr, m0, lane);
    __syncthreads();

    // -------- scores: (1/sqrt(128)) Q.K^T + mask bias --------
    for (int i = tid; i < ENVB * 64; i += THREADS) {
      int e = i >> 6, q = (i >> 3) & 7, g = i & 7;
      const _Float16* Qr = sQ + (e * 8 + q) * NH_;
      const _Float16* Kr = sK + (e * 8 + g) * NH_;
      float s = 0.0f;
      for (int d = 0; d < NH_; d += 8) {
        v8h qv = *(const v8h*)(Qr + d);
        v8h kv = *(const v8h*)(Kr + d);
#pragma unroll
        for (int j = 0; j < 8; ++j) s += (float)qv[j] * (float)kv[j];
      }
      sAttn[i] = s * 0.08838834764831845f + sBias[i];
    }
    __syncthreads();

    // -------- softmax over 8 neighbors, one thread per (env, q) row --------
    if (tid < ROWS) {
      float* row = sAttn + tid * 8;
      float mx = row[0];
#pragma unroll
      for (int g = 1; g < 8; ++g) mx = fmaxf(mx, row[g]);
      float ex[8], sum = 0.0f;
#pragma unroll
      for (int g = 0; g < 8; ++g) { ex[g] = __expf(row[g] - mx); sum += ex[g]; }
      float inv = 1.0f / sum;
#pragma unroll
      for (int g = 0; g < 8; ++g) row[g] = ex[g] * inv;   // masked lanes -> 0
    }
    __syncthreads();

    // -------- heads = attn @ V  -> overwrite sK (K is dead), v8h vectorized --------
    for (int i = tid; i < ROWS * NH_ / 8; i += THREADS) {
      int r = i >> 4;
      int d = (i & 15) * 8;
      int e = r >> 3;
      const float* arow = sAttn + r * 8;
      float acc[8] = {0, 0, 0, 0, 0, 0, 0, 0};
#pragma unroll
      for (int g = 0; g < 8; ++g) {
        v8h vv = *(const v8h*)(sV + (e * 8 + g) * NH_ + d);
        float w = arow[g];
#pragma unroll
        for (int j = 0; j < 8; ++j) acc[j] += w * (float)vv[j];
      }
      v8h hv;
#pragma unroll
      for (int j = 0; j < 8; ++j) hv[j] = (_Float16)acc[j];
      *(v8h*)(sK + r * NH_ + d) = hv;
    }
    __syncthreads();

    // -------- m = heads @ W_out (WMMA) -> sQ --------
    gemm_tile<NH_, NH_, NH_, NH_, NH_, false, false>(sK, sK, WoT, sQ, nullptr, m0, lane);
    __syncthreads();

    // -------- h' = relu(concat(h, m) @ upd_W + b) (WMMA, K=256 split) --------
    gemm_tile<256, NH_, NH_, 256, NH_, true, true>(hcur, sQ, updT, hnxt, sUpdB, m0, lane);
    __syncthreads();

    _Float16* t = hcur; hcur = hnxt; hnxt = t;
  }

  // -------- write output (A*BSZ, 128) f32, b128 stores --------
  for (int i = tid; i < ROWS * NH_ / 4; i += THREADS) {
    int r = i >> 5;
    int d = (i & 31) * 4;
    int e = r >> 3, a = r & 7;
    v4f o;
#pragma unroll
    for (int j = 0; j < 4; ++j) o[j] = (float)hcur[r * NH_ + d + j];
    *(v4f*)(out + ((size_t)a * NB + env0 + e) * NH_ + d) = o;
  }
}

extern "C" void kernel_launch(void* const* d_in, const int* in_sizes, int n_in,
                              void* d_out, int out_size, void* d_ws, size_t ws_size,
                              hipStream_t stream) {
  const float* inp  = (const float*)d_in[0];
  const float* encW = (const float*)d_in[1];
  const float* encb = (const float*)d_in[2];
  const float* Wq   = (const float*)d_in[3];
  const float* Wk   = (const float*)d_in[4];
  const float* Wv   = (const float*)d_in[5];
  const float* Wo   = (const float*)d_in[6];
  const float* updW = (const float*)d_in[7];
  const float* updb = (const float*)d_in[8];
  float* out = (float*)d_out;
  _Float16* ws = (_Float16*)d_ws;

  // 4096 + 4*16384 + 32768 = 102400 f16 elems of transposed weights
  prep_weights<<<(102400 + 255) / 256, 256, 0, stream>>>(encW, Wq, Wk, Wv, Wo, updW, ws);

  const _Float16* encT = ws;
  const _Float16* WqT  = ws + 128 * 32;
  const _Float16* WkT  = WqT + 128 * 128;
  const _Float16* WvT  = WkT + 128 * 128;
  const _Float16* WoT  = WvT + 128 * 128;
  const _Float16* updT = WoT + 128 * 128;

  size_t smem = (size_t)5 * ROWS * NH_ * sizeof(_Float16)              // h0,h1,Q,K,V
              + (size_t)(2 * ENVB * 64 + 2 * NH_ + ENVB * NA * 2) * sizeof(float);
  mpnn_fused<<<NB / ENVB, THREADS, smem, stream>>>(inp, encb, updb,
                                                   encT, WqT, WkT, WvT, WoT, updT, out);
}
